// Attention_19481971655342
// MI455X (gfx1250) — compile-verified, matrix operand order checked by reading, not run on hardware
//
#include <hip/hip_runtime.h>
#include <hip/hip_bf16.h>
#include <math.h>

// ---------------- problem constants ----------------
#define BTRUE  2
#define SDIM   2
#define BS     4          // BTRUE * SDIM
#define NSEQ   1024
#define DIM    512
#define HEADS  8
#define DHEAD  64
#define DINNER 512        // HEADS * DHEAD
#define AH     16         // SDIM * HEADS (talking-heads channels)
#define NEG_MAX 3.402823466e38f

typedef __attribute__((ext_vector_type(16))) __bf16 v16bf;
typedef __attribute__((ext_vector_type(8)))  float  v8f;
typedef __attribute__((ext_vector_type(4)))  unsigned int u32x4;
typedef __attribute__((ext_vector_type(8)))  int  i32x8;
typedef __attribute__((ext_vector_type(4)))  int  i32x4;

// ---------------- helpers ----------------
__device__ __forceinline__ unsigned short f2bf_bits(float f) {
  unsigned int u = __builtin_bit_cast(unsigned int, f);
  unsigned int r = u + 0x7FFFu + ((u >> 16) & 1u);   // round-to-nearest-even
  return (unsigned short)(r >> 16);
}

// A fragment: 16x32 (MxK), row-major source, lda in elements.
// Lanes 0-15: rows M=lane, K {0..7, 16..23}; lanes 16-31: K {8..15, 24..31}.
// Per lane: two contiguous 16B chunks -> b128 pairs.
__device__ __forceinline__ v16bf load_a_bf16(const unsigned short* base, int lda) {
  int lane = threadIdx.x & 31;
  int half = lane >> 4, row = lane & 15;
  const unsigned short* p = base + (size_t)row * lda;
  v16bf a;
#pragma unroll
  for (int vi = 0; vi < 8; ++vi) {
    int kk = ((vi & 4) ? (16 + 2 * (vi & 3)) : (2 * vi)) + (half << 3);
    a[2 * vi]     = __builtin_bit_cast(__bf16, p[kk]);
    a[2 * vi + 1] = __builtin_bit_cast(__bf16, p[kk + 1]);
  }
  return a;
}

// B fragment: 32x16 (KxN), element (k,n) at base[k*kStride + n*nStride]; lanes = N cols.
// With kStride==1 this is per-lane contiguous -> b128 pairs.
__device__ __forceinline__ v16bf load_b_bf16(const unsigned short* base, int kStride, int nStride) {
  int lane = threadIdx.x & 31;
  int half = lane >> 4, col = lane & 15;
  const unsigned short* p = base + (size_t)col * nStride;
  v16bf b;
#pragma unroll
  for (int vi = 0; vi < 8; ++vi) {
    int kk = ((vi & 4) ? (16 + 2 * (vi & 3)) : (2 * vi)) + (half << 3);
    b[2 * vi]     = __builtin_bit_cast(__bf16, p[(size_t)kk * kStride]);
    b[2 * vi + 1] = __builtin_bit_cast(__bf16, p[(size_t)(kk + 1) * kStride]);
  }
  return b;
}

__device__ __forceinline__ v8f wmma_bf16(v16bf a, v16bf b, v8f c) {
  return __builtin_amdgcn_wmma_f32_16x16x32_bf16(false, a, false, b, (short)0, c, false, false);
}

// ---------------- Tensor Data Mover staging (global u16 tile -> LDS) ----------------
#if __has_builtin(__builtin_amdgcn_tensor_load_to_lds)
#define HAS_TDM 1
#if __has_include(<hip/amd_detail/amd_gfx1250_TDM.h>)
#warning "CDNA5 probe: TDM builtin available, using 6-arg form"
#else
#warning "CDNA5 probe: TDM builtin available, using 5-arg form"
#endif
#else
#warning "CDNA5 probe: TDM builtin NOT available, using cooperative-copy staging"
#endif

#ifdef HAS_TDM
// 2-D tile load: rows x rowElems (u16), tensor_dim0_stride == rowElems (contiguous tile).
__device__ __forceinline__ void tdm_issue_load(unsigned ldsOff, const void* gsrc,
                                               unsigned rowElems, unsigned rows) {
  unsigned long long ga = (unsigned long long)gsrc;
  unsigned long long s0 = rowElems;
  unsigned long long s1 = (unsigned long long)rowElems * rows;
  u32x4 g0;
  g0.x = 1u;                                    // count=1 (valid user descriptor)
  g0.y = ldsOff;                                // lds_addr
  g0.z = (unsigned)ga;                          // global_addr[31:0]
  g0.w = (unsigned)(ga >> 32) | (2u << 30);     // global_addr[56:32] | type=2
  i32x8 g1;
  g1[0] = (int)(1u << 16);                                              // data_size=2B
  g1[1] = (int)((rowElems & 0xFFFFu) << 16);                            // tensor_dim0 lo16
  g1[2] = (int)((rowElems >> 16) | ((rows & 0xFFFFu) << 16));           // dim0 hi | dim1 lo
  g1[3] = (int)((rows >> 16) | ((rowElems & 0xFFFFu) << 16));           // dim1 hi | tile_dim0
  g1[4] = (int)(rows & 0xFFFFu);                                        // tile_dim1 (tile_dim2=0)
  g1[5] = (int)(unsigned)(s0 & 0xFFFFFFFFu);                            // dim0_stride lo32
  g1[6] = (int)((unsigned)(s0 >> 32) | ((unsigned)(s1 & 0xFFFFu) << 16)); // str0 hi | str1 lo
  g1[7] = (int)(unsigned)((s1 >> 16) & 0xFFFFFFFFu);                    // str1[47:16]
  i32x4 z4 = {0, 0, 0, 0};
#if __has_include(<hip/amd_detail/amd_gfx1250_TDM.h>)
  i32x8 z8 = {0, 0, 0, 0, 0, 0, 0, 0};
  __builtin_amdgcn_tensor_load_to_lds(g0, g1, z4, z4, z8, 0);           // 6-arg toolchain
#else
  __builtin_amdgcn_tensor_load_to_lds(g0, g1, z4, z4, 0);               // 5-arg toolchain
#endif
}
#endif

__device__ __forceinline__ void stage_tile_u16(unsigned short* dst, const unsigned short* src,
                                               unsigned rowElems, unsigned rows) {
#ifdef HAS_TDM
  if (threadIdx.x < 32) {                       // one wave issues the DMA
    tdm_issue_load((unsigned)(size_t)(void*)dst, (const void*)src, rowElems, rows);
    __builtin_amdgcn_s_wait_tensorcnt(0);
  }
  __syncthreads();
#else
  const uint4* s4 = (const uint4*)src;
  uint4* d4 = (uint4*)dst;
  unsigned total = (rowElems * rows) >> 3;      // u16 -> uint4 (8 elems)
  for (unsigned i = threadIdx.x; i < total; i += blockDim.x) d4[i] = s4[i];
  __syncthreads();
#endif
}

// ---------------- K0: fp32 -> bf16 weight conversion ----------------
__global__ void k_f32_to_bf16(const float* __restrict__ src, unsigned short* __restrict__ dst, int n) {
  int i = blockIdx.x * blockDim.x + threadIdx.x;
  if (i < n) dst[i] = f2bf_bits(src[i]);
}

// ---------------- K1: RMSNorm + sigmoid gates (one block per token row) ----------------
__global__ void k_rmsnorm_gates(const float* __restrict__ x, const float* __restrict__ gamma,
                                const float* __restrict__ w_gates, const float* __restrict__ b_gates,
                                unsigned short* __restrict__ xn, float* __restrict__ gates) {
  __shared__ float red[256];
  int row = blockIdx.x;
  int t = threadIdx.x;
  const float* xr = x + (size_t)row * DIM;
  float v0 = xr[t], v1 = xr[t + 256];
  red[t] = v0 * v0 + v1 * v1;
  __syncthreads();
  for (int s2 = 128; s2 > 0; s2 >>= 1) {
    if (t < s2) red[t] += red[t + s2];
    __syncthreads();
  }
  float rs = rsqrtf(red[0] / (float)DIM + 1e-5f);
  __syncthreads();
  float n0 = v0 * rs * gamma[t];
  float n1 = v1 * rs * gamma[t + 256];
  xn[(size_t)row * DIM + t]       = f2bf_bits(n0);
  xn[(size_t)row * DIM + t + 256] = f2bf_bits(n1);
  int bhat = row / NSEQ, n = row % NSEQ;
  for (int h = 0; h < HEADS; ++h) {
    const float* wg = w_gates + (size_t)h * DIM;
    red[t] = n0 * wg[t] + n1 * wg[t + 256];
    __syncthreads();
    for (int s2 = 128; s2 > 0; s2 >>= 1) {
      if (t < s2) red[t] += red[t + s2];
      __syncthreads();
    }
    if (t == 0) {
      float g = red[0] + b_gates[h];
      gates[((size_t)bhat * HEADS + h) * NSEQ + n] = 1.0f / (1.0f + __expf(-g));
    }
    __syncthreads();
  }
}

// ---------------- K2: QKV projection (128x128 block tile, weight slice staged in LDS) ----------------
// grid (12, 32). Dyn LDS 128KB. q/k stored [bh][n][d]; V stored TRANSPOSED [bh][d][n] for K5.
__global__ void __launch_bounds__(256, 1)
k_qkv_gemm(const unsigned short* __restrict__ xn, const unsigned short* __restrict__ wqkv,
           unsigned short* __restrict__ qb, unsigned short* __restrict__ kb,
           unsigned short* __restrict__ vbT) {
  extern __shared__ unsigned short wt2[];       // [128][DIM]
  int t = threadIdx.x;
  int colBlock = blockIdx.x, rowBlock = blockIdx.y;
  stage_tile_u16(wt2, wqkv + (size_t)colBlock * 128 * DIM, DIM, 128);
  int wave = t >> 5, lane = t & 31;
  int row0 = rowBlock * 128 + wave * 16;
  v8f acc[8] = {};
  for (int k = 0; k < DIM; k += 32) {
    v16bf a = load_a_bf16(xn + (size_t)row0 * DIM + k, DIM);
#pragma unroll
    for (int j = 0; j < 8; ++j) {
      v16bf b = load_b_bf16(wt2 + (size_t)(j * 16) * DIM + k, 1, DIM);
      acc[j] = wmma_bf16(a, b, acc[j]);
    }
  }
  int col = lane & 15, hw = lane >> 4;
#pragma unroll
  for (int j = 0; j < 8; ++j) {
    int e = colBlock * 128 + j * 16 + col;
    int which = e >> 9;          // / DINNER
    int rem = e & 511;           // % DINNER
    int h = rem >> 6, d = rem & 63;
#pragma unroll
    for (int r = 0; r < 8; ++r) {
      int m = row0 + r + 8 * hw;
      int bhat = m >> 10, n = m & 1023;
      float val = acc[j][r];
      if (which == 0) {
        qb[(((size_t)bhat * HEADS + h) * NSEQ + n) * DHEAD + d] = f2bf_bits(val * 8.0f); // * sqrt(Dh)
      } else if (which == 1) {
        kb[(((size_t)bhat * HEADS + h) * NSEQ + n) * DHEAD + d] = f2bf_bits(val);
      } else {
        vbT[(((size_t)bhat * HEADS + h) * DHEAD + d) * NSEQ + n] = f2bf_bits(val); // transposed
      }
    }
  }
}

// ---------------- K3: raw sim = Q.K^T, 128x128 block tiles, Q/K tiles staged in LDS ----------------
// grid (8, 8, 32): (jblk, iblk, bh). Static LDS 32KB.
__global__ void __launch_bounds__(256, 1)
k_sim(const unsigned short* __restrict__ qb, const unsigned short* __restrict__ kb,
      float* __restrict__ simout) {
  __shared__ unsigned short qt[128 * DHEAD];
  __shared__ unsigned short kt[128 * DHEAD];
  int t = threadIdx.x;
  int bh = blockIdx.z;
  int i0 = blockIdx.y * 128, j0 = blockIdx.x * 128;
  stage_tile_u16(qt, qb + ((size_t)bh * NSEQ + i0) * DHEAD, DHEAD, 128);
  stage_tile_u16(kt, kb + ((size_t)bh * NSEQ + j0) * DHEAD, DHEAD, 128);
  int wave = t >> 5, lane = t & 31;
  v8f acc[8] = {};
  for (int k = 0; k < DHEAD; k += 32) {
    v16bf a = load_a_bf16(qt + (size_t)(wave * 16) * DHEAD + k, DHEAD);
#pragma unroll
    for (int j = 0; j < 8; ++j) {
      v16bf b = load_b_bf16(kt + (size_t)(j * 16) * DHEAD + k, 1, DHEAD); // B(k,n)=K[j0+n][k]
      acc[j] = wmma_bf16(a, b, acc[j]);
    }
  }
  int col = lane & 15, hw = lane >> 4;
#pragma unroll
  for (int j = 0; j < 8; ++j)
#pragma unroll
    for (int r = 0; r < 8; ++r) {
      size_t idx = ((size_t)bh * NSEQ + i0 + wave * 16 + r + 8 * hw) * NSEQ + j0 + j * 16 + col;
      simout[idx] = acc[j][r];
    }
}

// ---------------- K4: pre-mix + mask + causal + softmax (in-place output) + post-mix -> bf16 ----------
// grid (1024, 2) = (row i, b). Dyn LDS 128KB: raw[16][1024] + prob[16][1024].
__global__ void __launch_bounds__(256, 1)
k_softmax_mix(float* __restrict__ attn, const unsigned char* __restrict__ mask,
              const float* __restrict__ w_pre, const float* __restrict__ w_post,
              unsigned short* __restrict__ attn2) {
  extern __shared__ float sm4[];
  float* raw  = sm4;                 // [AH][NSEQ]
  float* prob = sm4 + AH * NSEQ;     // [AH][NSEQ]
  __shared__ float red[8];
  __shared__ float wpre[AH * AH], wpost[AH * AH];
  int i = blockIdx.x, b = blockIdx.y, t = threadIdx.x;
  int lane = t & 31, wave = t >> 5;
  wpre[t]  = w_pre[t];
  wpost[t] = w_post[t];
  for (int c = 0; c < AH; ++c) {
    const float* grow = attn + (((size_t)(b * SDIM + c / HEADS) * HEADS + (c % HEADS)) * NSEQ + i) * NSEQ;
#pragma unroll
    for (int kk = 0; kk < 4; ++kk) raw[c * NSEQ + t + kk * 256] = grow[t + kk * 256];
  }
  __syncthreads();
  for (int c = 0; c < AH; ++c) {
    int bhat = b * SDIM + c / HEADS, h = c % HEADS;
    const unsigned char* mrow = mask + (size_t)bhat * NSEQ;
    float v[4];
    float mx = -NEG_MAX;
#pragma unroll
    for (int kk = 0; kk < 4; ++kk) {
      int j = t + kk * 256;
      float pv = 0.f;                             // pre talking-heads mix
#pragma unroll
      for (int ci = 0; ci < AH; ++ci) pv += wpre[c * AH + ci] * raw[ci * NSEQ + j];
      if (j > i || !mrow[j]) pv = -NEG_MAX;       // causal + key-padding mask
      v[kk] = pv;
      mx = fmaxf(mx, pv);
    }
    for (int off = 16; off > 0; off >>= 1) mx = fmaxf(mx, __shfl_xor(mx, off, 32));
    if (lane == 0) red[wave] = mx;
    __syncthreads();
    if (t == 0) { float m = red[0]; for (int w = 1; w < 8; ++w) m = fmaxf(m, red[w]); red[0] = m; }
    __syncthreads();
    mx = red[0];
    __syncthreads();
    float sum = 0.f;
#pragma unroll
    for (int kk = 0; kk < 4; ++kk) { float e = __expf(v[kk] - mx); v[kk] = e; sum += e; }
    for (int off = 16; off > 0; off >>= 1) sum += __shfl_xor(sum, off, 32);
    if (lane == 0) red[wave] = sum;
    __syncthreads();
    if (t == 0) { float s2 = 0.f; for (int w = 0; w < 8; ++w) s2 += red[w]; red[0] = s2; }
    __syncthreads();
    float inv = 1.0f / red[0];
    float* grow = attn + (((size_t)bhat * HEADS + h) * NSEQ + i) * NSEQ;
#pragma unroll
    for (int kk = 0; kk < 4; ++kk) {
      int j = t + kk * 256;
      float p = v[kk] * inv;
      grow[j] = p;                                // post_softmax_attn output
      prob[c * NSEQ + j] = p;
    }
    __syncthreads();
  }
  // post talking-heads mix -> bf16 attn2 for A.V GEMM
  for (int co = 0; co < AH; ++co) {
    int so = co / HEADS, ho = co % HEADS;
    size_t base = (((size_t)(b * SDIM + so) * HEADS + ho) * NSEQ + i) * NSEQ;
#pragma unroll
    for (int kk = 0; kk < 4; ++kk) {
      int j = t + kk * 256;
      float acc = 0.f;
#pragma unroll
      for (int ci = 0; ci < AH; ++ci) acc += wpost[co * AH + ci] * prob[ci * NSEQ + j];
      attn2[base + j] = f2bf_bits(acc);
    }
  }
}

// ---------------- K5: attn2.V + gating; whole V^T[bh] (128KB) staged in LDS via TDM ----------------
// grid (8, 32). Dyn LDS 128KB. vbT is [bh][d][k] so B fragments are per-lane contiguous.
__global__ void __launch_bounds__(256, 1)
k_av_gate(const unsigned short* __restrict__ attn2, const unsigned short* __restrict__ vbT,
          const float* __restrict__ gates, unsigned short* __restrict__ hidden) {
  extern __shared__ unsigned short vt[];         // [DHEAD][NSEQ] (transposed V)
  int t = threadIdx.x;
  int bh = blockIdx.y, bhat = bh >> 3, h = bh & 7;
  stage_tile_u16(vt, vbT + (size_t)bh * DHEAD * NSEQ, NSEQ, DHEAD);
  int wave = t >> 5, lane = t & 31;
  int i0 = blockIdx.x * 128 + wave * 16;
  const unsigned short* abase = attn2 + ((size_t)bh * NSEQ + i0) * NSEQ;
  v8f acc[4] = {};
  for (int k = 0; k < NSEQ; k += 32) {
    __builtin_prefetch(abase + k + 64, 0, 0);    // global_prefetch_b8 of next A chunk
    v16bf a = load_a_bf16(abase + k, NSEQ);
#pragma unroll
    for (int dt = 0; dt < 4; ++dt) {
      // B(k,n) = vt[dt*16+n][k] : kStride=1 (contiguous), nStride=NSEQ
      v16bf b = load_b_bf16(vt + (size_t)(dt * 16) * NSEQ + k, 1, NSEQ);
      acc[dt] = wmma_bf16(a, b, acc[dt]);
    }
  }
  int col = lane & 15, hw = lane >> 4;
#pragma unroll
  for (int dt = 0; dt < 4; ++dt)
#pragma unroll
    for (int r = 0; r < 8; ++r) {
      int m = i0 + r + 8 * hw;
      float g = gates[((size_t)bhat * HEADS + h) * NSEQ + m];
      size_t idx = ((size_t)bhat * NSEQ + m) * DINNER + (size_t)h * DHEAD + dt * 16 + col;
      hidden[idx] = f2bf_bits(acc[dt][r] * g);
    }
}

// ---------------- K6: output projection, 128x128 block tiles, w_out slice staged in LDS ----------------
// grid (4, 32). Dyn LDS 128KB.
__global__ void __launch_bounds__(256, 1)
k_out_gemm(const unsigned short* __restrict__ hidden, const unsigned short* __restrict__ wout,
           float* __restrict__ out) {
  extern __shared__ unsigned short wt6[];        // [128][DINNER]
  int t = threadIdx.x;
  int colBlock = blockIdx.x, rowBlock = blockIdx.y;
  stage_tile_u16(wt6, wout + (size_t)colBlock * 128 * DINNER, DINNER, 128);
  int wave = t >> 5, lane = t & 31;
  int row0 = rowBlock * 128 + wave * 16;
  v8f acc[8] = {};
  for (int k = 0; k < DINNER; k += 32) {
    v16bf a = load_a_bf16(hidden + (size_t)row0 * DINNER + k, DINNER);
#pragma unroll
    for (int j = 0; j < 8; ++j) {
      v16bf b = load_b_bf16(wt6 + (size_t)(j * 16) * DINNER + k, 1, DINNER);
      acc[j] = wmma_bf16(a, b, acc[j]);
    }
  }
  int col = lane & 15, hw = lane >> 4;
#pragma unroll
  for (int j = 0; j < 8; ++j)
#pragma unroll
    for (int r = 0; r < 8; ++r)
      out[(size_t)(row0 + r + 8 * hw) * DIM + colBlock * 128 + j * 16 + col] = acc[j][r];
}

// ---------------- host launch ----------------
extern "C" void kernel_launch(void* const* d_in, const int* in_sizes, int n_in,
                              void* d_out, int out_size, void* d_ws, size_t ws_size,
                              hipStream_t stream) {
  (void)in_sizes; (void)n_in; (void)out_size; (void)ws_size;
  const float*         x       = (const float*)d_in[0];
  const unsigned char* mask    = (const unsigned char*)d_in[1];   // bool
  const float*         gamma   = (const float*)d_in[2];
  const float*         w_qkv   = (const float*)d_in[3];
  const float*         w_gates = (const float*)d_in[4];
  const float*         b_gates = (const float*)d_in[5];
  const float*         w_pre   = (const float*)d_in[6];
  const float*         w_post  = (const float*)d_in[7];
  const float*         w_out   = (const float*)d_in[8];

  float* out_proj = (float*)d_out;                                // [BS, N, DIM]
  float* attn_out = out_proj + (size_t)BS * NSEQ * DIM;           // [BS, H, N, N]

  char* ws = (char*)d_ws;
  size_t off = 0;
  auto carve = [&](size_t bytes) { char* p = ws + off; off = (off + bytes + 255) & ~(size_t)255; return p; };
  unsigned short* xn      = (unsigned short*)carve((size_t)BS * NSEQ * DIM * 2);
  unsigned short* wqkv_bf = (unsigned short*)carve((size_t)3 * DINNER * DIM * 2);
  unsigned short* wout_bf = (unsigned short*)carve((size_t)DIM * DINNER * 2);
  unsigned short* qb      = (unsigned short*)carve((size_t)BS * HEADS * NSEQ * DHEAD * 2);
  unsigned short* kb      = (unsigned short*)carve((size_t)BS * HEADS * NSEQ * DHEAD * 2);
  unsigned short* vbT     = (unsigned short*)carve((size_t)BS * HEADS * DHEAD * NSEQ * 2);
  float*          gates   = (float*)carve((size_t)BS * HEADS * NSEQ * 4);
  unsigned short* attn2   = (unsigned short*)carve((size_t)BS * HEADS * NSEQ * NSEQ * 2);
  unsigned short* hidden  = (unsigned short*)carve((size_t)BS * NSEQ * DINNER * 2);

  const int nqkv = 3 * DINNER * DIM, nout = DIM * DINNER;
  const size_t LDS128K = 128u * 1024u;
  k_f32_to_bf16<<<(nqkv + 255) / 256, 256, 0, stream>>>(w_qkv, wqkv_bf, nqkv);
  k_f32_to_bf16<<<(nout + 255) / 256, 256, 0, stream>>>(w_out, wout_bf, nout);
  k_rmsnorm_gates<<<BS * NSEQ, 256, 0, stream>>>(x, gamma, w_gates, b_gates, xn, gates);
  k_qkv_gemm<<<dim3(12, 32), 256, LDS128K, stream>>>(xn, wqkv_bf, qb, kb, vbT);
  k_sim<<<dim3(NSEQ / 128, NSEQ / 128, BS * HEADS), 256, 0, stream>>>(qb, kb, attn_out);
  k_softmax_mix<<<dim3(NSEQ, BTRUE), 256, LDS128K, stream>>>(attn_out, mask, w_pre, w_post, attn2);
  k_av_gate<<<dim3(NSEQ / 128, BS * HEADS), 256, LDS128K, stream>>>(attn2, vbT, gates, hidden);
  k_out_gemm<<<dim3(4, 32), 256, LDS128K, stream>>>(hidden, wout_bf, out_proj);
}
